// BaseTransformerLayer_27187142984229
// MI455X (gfx1250) — matrix operand order repaired and checked by reading, not compiled, not run on hardware
//
#include <hip/hip_runtime.h>

// Problem dims
#define Bn  4
#define Sn  1024
#define Hn  2048
#define NHn 32
#define HDn 64
#define Mn  (Bn * Sn)   // 4096 rows
#define H3n (3 * Hn)    // 6144
#define H4n (4 * Hn)    // 8192

typedef _Float16 half_t;
typedef __attribute__((ext_vector_type(16))) _Float16 v16h;
typedef __attribute__((ext_vector_type(8)))  _Float16 v8h;
typedef __attribute__((ext_vector_type(8)))  float    v8f;

// gfx1250 async global->LDS path (graceful fallback if builtins absent)
#if __has_builtin(__builtin_amdgcn_global_load_async_to_lds_b128) && \
    __has_builtin(__builtin_amdgcn_s_wait_asynccnt)
#define USE_ASYNC_LDS 1
#else
#define USE_ASYNC_LDS 0
#endif

#if USE_ASYNC_LDS
// clang signature: void(v4i addrspace(1)*, v4i addrspace(3)*, imm int, imm int)
typedef __attribute__((__vector_size__(4 * sizeof(int)))) int v4i_b128;
#define ASYNC_CP_B128(gp, lp)                                                   \
  __builtin_amdgcn_global_load_async_to_lds_b128(                               \
      (__attribute__((address_space(1))) v4i_b128*)(gp),                        \
      (__attribute__((address_space(3))) v4i_b128*)(lp), 0, 0)
#define ASYNC_WAIT() __builtin_amdgcn_s_wait_asynccnt(0)
#else
#define ASYNC_WAIT()
#endif

union U64h { uint2 u; half_t h[4]; };

// ---------------------------------------------------------------------------
// WMMA helper: D = A(16x32 f16) * B(32x16 f16) + C(16x16 f32)
// ---------------------------------------------------------------------------
static __device__ __forceinline__ v8f wmma16x16x32(v16h a, v16h b, v8f c) {
  return __builtin_amdgcn_wmma_f32_16x16x32_f16(false, a, false, b, (short)0, c,
                                                false, false);
}

// A-fragment (16x32) from row-major LDS tile [m][k]. stride %8==0, koff %32==0.
// ISA layout: lane L holds row M=L&15; half e -> K=(e&7)+8*(L>>4)+16*(e>>3).
static __device__ __forceinline__ v16h frag_a_lds(const half_t* base, int mbase,
                                                  int stride, int lane, int koff) {
  const half_t* p =
      base + (size_t)(mbase + (lane & 15)) * stride + 8 * (lane >> 4) + koff;
  v8h lo = *(const v8h*)(p);        // K = koff + 8*hl .. +7
  v8h hi = *(const v8h*)(p + 16);   // K = koff + 16 + 8*hl .. +7
  return __builtin_shufflevector(lo, hi, 0, 1, 2, 3, 4, 5, 6, 7,
                                 8, 9, 10, 11, 12, 13, 14, 15);
}

// B-fragment (32x16) from LDS tile stored [n][k] (k contiguous). stride %8==0,
// koff %32==0. ISA layout: lane L holds col N=L&15; half e -> K=(e&7)+8*(L>>4)+16*(e>>3).
static __device__ __forceinline__ v16h frag_b_lds(const half_t* base, int nbase,
                                                  int stride, int lane, int koff) {
  const half_t* p =
      base + (size_t)(nbase + (lane & 15)) * stride + 8 * (lane >> 4) + koff;
  v8h lo = *(const v8h*)(p);
  v8h hi = *(const v8h*)(p + 16);
  return __builtin_shufflevector(lo, hi, 0, 1, 2, 3, 4, 5, 6, 7,
                                 8, 9, 10, 11, 12, 13, 14, 15);
}

// fast gelu(tanh approx) without libm tanh's branchy path
static __device__ __forceinline__ float gelu_f(float x) {
  float u = 0.7978845608028654f * (x + 0.044715f * x * x * x);
  float z = fminf(fmaxf(2.f * u, -40.f), 40.f);
  float e = __expf(z);
  float th = (e - 1.f) / (e + 1.f);
  return 0.5f * x * (1.f + th);
}

// ---------------------------------------------------------------------------
// fp32 -> f16 conversion (weights)
// ---------------------------------------------------------------------------
__global__ __launch_bounds__(256) void cvt_f32_f16(const float* __restrict__ in,
                                                   half_t* __restrict__ out, int n) {
  int i = blockIdx.x * blockDim.x + threadIdx.x;
  int stride = gridDim.x * blockDim.x;
  for (; i < n; i += stride) out[i] = (half_t)in[i];
}

// ---------------------------------------------------------------------------
// Block reduction helper (256 threads): sums rs[] and rs2[]
// ---------------------------------------------------------------------------
static __device__ __forceinline__ void block_reduce_2(float* rs, float* rs2, int tid) {
  __syncthreads();
  for (int off = 128; off > 0; off >>= 1) {
    if (tid < off) {
      rs[tid]  += rs[tid + off];
      rs2[tid] += rs2[tid + off];
    }
    __syncthreads();
  }
}

// ---------------------------------------------------------------------------
// LN1: x1 = f16(LayerNorm(hidden))
// ---------------------------------------------------------------------------
__global__ __launch_bounds__(256) void ln_to_f16(const float* __restrict__ x,
                                                 const float* __restrict__ wt,
                                                 const float* __restrict__ bs,
                                                 half_t* __restrict__ out) {
  int row = blockIdx.x, tid = threadIdx.x;
  const float* xr = x + (size_t)row * Hn;
  __shared__ float rs[256], rs2[256];
  float v[8], s = 0.f, s2 = 0.f;
#pragma unroll
  for (int i = 0; i < 8; i++) {
    float t = xr[tid + 256 * i];
    v[i] = t; s += t; s2 += t * t;
  }
  rs[tid] = s; rs2[tid] = s2;
  block_reduce_2(rs, rs2, tid);
  float mu = rs[0] * (1.f / Hn);
  float var = rs2[0] * (1.f / Hn) - mu * mu;
  float rstd = rsqrtf(var + 1e-5f);
#pragma unroll
  for (int i = 0; i < 8; i++) {
    int c = tid + 256 * i;
    out[(size_t)row * Hn + c] = (half_t)((v[i] - mu) * rstd * wt[c] + bs[c]);
  }
}

// ---------------------------------------------------------------------------
// Fused: ln_input = hidden + LN3(attn);  x2 = f16(LN2(ln_input))
// ---------------------------------------------------------------------------
__global__ __launch_bounds__(256) void fuse_ln3_res_ln2(
    const float* __restrict__ attn, const float* __restrict__ hid,
    const float* __restrict__ w3, const float* __restrict__ b3,
    const float* __restrict__ w2, const float* __restrict__ b2,
    float* __restrict__ lnin, half_t* __restrict__ x2) {
  int row = blockIdx.x, tid = threadIdx.x;
  const float* ar = attn + (size_t)row * Hn;
  const float* hr = hid + (size_t)row * Hn;
  __shared__ float rs[256], rs2[256];
  float a[8], s = 0.f, s2 = 0.f;
#pragma unroll
  for (int i = 0; i < 8; i++) {
    float t = ar[tid + 256 * i];
    a[i] = t; s += t; s2 += t * t;
  }
  rs[tid] = s; rs2[tid] = s2;
  block_reduce_2(rs, rs2, tid);
  float mu = rs[0] * (1.f / Hn);
  float var = rs2[0] * (1.f / Hn) - mu * mu;
  float rstd = rsqrtf(var + 1e-5f);
  __syncthreads();  // rs reuse

  float t8[8]; s = 0.f; s2 = 0.f;
#pragma unroll
  for (int i = 0; i < 8; i++) {
    int c = tid + 256 * i;
    float x = hr[c] + ((a[i] - mu) * rstd * w3[c] + b3[c]);
    t8[i] = x;
    lnin[(size_t)row * Hn + c] = x;
    s += x; s2 += x * x;
  }
  rs[tid] = s; rs2[tid] = s2;
  block_reduce_2(rs, rs2, tid);
  float mu2 = rs[0] * (1.f / Hn);
  float var2 = rs2[0] * (1.f / Hn) - mu2 * mu2;
  float rstd2 = rsqrtf(var2 + 1e-5f);
#pragma unroll
  for (int i = 0; i < 8; i++) {
    int c = tid + 256 * i;
    x2[(size_t)row * Hn + c] = (half_t)((t8[i] - mu2) * rstd2 * w2[c] + b2[c]);
  }
}

// ---------------------------------------------------------------------------
// Final: out = ln_input + LN4(mlp)
// ---------------------------------------------------------------------------
__global__ __launch_bounds__(256) void final_ln4_res(const float* __restrict__ mlp,
                                                     const float* __restrict__ lnin,
                                                     const float* __restrict__ w4,
                                                     const float* __restrict__ b4,
                                                     float* __restrict__ out) {
  int row = blockIdx.x, tid = threadIdx.x;
  const float* mr = mlp + (size_t)row * Hn;
  const float* lr = lnin + (size_t)row * Hn;
  __shared__ float rs[256], rs2[256];
  float v[8], s = 0.f, s2 = 0.f;
#pragma unroll
  for (int i = 0; i < 8; i++) {
    float t = mr[tid + 256 * i];
    v[i] = t; s += t; s2 += t * t;
  }
  rs[tid] = s; rs2[tid] = s2;
  block_reduce_2(rs, rs2, tid);
  float mu = rs[0] * (1.f / Hn);
  float var = rs2[0] * (1.f / Hn) - mu * mu;
  float rstd = rsqrtf(var + 1e-5f);
#pragma unroll
  for (int i = 0; i < 8; i++) {
    int c = tid + 256 * i;
    out[(size_t)row * Hn + c] = lr[c] + ((v[i] - mu) * rstd * w4[c] + b4[c]);
  }
}

// ---------------------------------------------------------------------------
// Tiled WMMA GEMM: C[M,N] = A[M,K](f16) * B[K,N](f16) + bias
// Block: 256 thr (8 waves), tile 128x128, wave 32x64, K-stage 64.
// mode 0: fp32 store | mode 1: gelu -> f16 store | mode 2: scatter to q/k/v
// ---------------------------------------------------------------------------
__global__ __launch_bounds__(256) void gemm_f16_kernel(
    const half_t* __restrict__ A, const half_t* __restrict__ Bw,
    const float* __restrict__ bias, int Kdim, int Ndim, int mode,
    float* __restrict__ outF, half_t* __restrict__ outH,
    half_t* __restrict__ qo, half_t* __restrict__ ko, half_t* __restrict__ vo) {
  __shared__ __align__(16) half_t As[128][72];  // [m][k], stride 72 -> b128 ok
  __shared__ __align__(16) half_t Bs[128][72];  // [n][k] transposed
  const int tid = threadIdx.x;
  const int lane = tid & 31;
  const int w = tid >> 5;
  const int bm = blockIdx.y * 128;
  const int bn = blockIdx.x * 128;
  const int wm = (w >> 1) * 32;   // 4 waves down
  const int wn = (w & 1) * 64;    // 2 waves across

  v8f acc[2][4] = {};

  for (int k0 = 0; k0 < Kdim; k0 += 64) {
    __syncthreads();
    // prefetch next K-stage (global_prefetch_b8)
    if (k0 + 64 < Kdim) {
      __builtin_prefetch((const void*)(A + (size_t)(bm + (tid >> 1)) * Kdim + k0 + 64 +
                                       ((tid & 1) << 5)), 0, 1);
      __builtin_prefetch((const void*)(Bw + (size_t)(k0 + 64 + (tid >> 2)) * Ndim + bn +
                                       ((tid & 3) << 5)), 0, 1);
    }
    // ---- A tile 128x64, row-major copy ----
#if USE_ASYNC_LDS
#pragma unroll
    for (int i = 0; i < 4; i++) {
      int idx = i * 256 + tid;          // 1024 x 16B chunks
      int r = idx >> 3;                 // row 0..127
      int c8 = (idx & 7) << 3;          // half col 0..56
      ASYNC_CP_B128(A + (size_t)(bm + r) * Kdim + k0 + c8, &As[r][c8]);
    }
#else
#pragma unroll
    for (int i = 0; i < 8; i++) {
      int idx = i * 256 + tid;          // 2048 x uint2
      int r = idx >> 4;
      int c4 = (idx & 15) << 2;
      uint2 d = *(const uint2*)(A + (size_t)(bm + r) * Kdim + k0 + c4);
      *(uint2*)(&As[r][c4]) = d;
    }
#endif
    // ---- B tile 64x128 -> transposed [n][k] via 4x4 register transpose ----
#pragma unroll
    for (int i = 0; i < 2; i++) {
      int sb = i * 256 + tid;           // 512 sub-blocks of 4k x 4n
      int kb = (sb >> 5) << 2;          // 0..60
      int nb = (sb & 31) << 2;          // 0..124
      uint2 d0 = *(const uint2*)(Bw + (size_t)(k0 + kb + 0) * Ndim + bn + nb);
      uint2 d1 = *(const uint2*)(Bw + (size_t)(k0 + kb + 1) * Ndim + bn + nb);
      uint2 d2 = *(const uint2*)(Bw + (size_t)(k0 + kb + 2) * Ndim + bn + nb);
      uint2 d3 = *(const uint2*)(Bw + (size_t)(k0 + kb + 3) * Ndim + bn + nb);
      const half_t* h0 = (const half_t*)&d0;
      const half_t* h1 = (const half_t*)&d1;
      const half_t* h2 = (const half_t*)&d2;
      const half_t* h3 = (const half_t*)&d3;
#pragma unroll
      for (int j = 0; j < 4; j++) {
        U64h t;
        t.h[0] = h0[j]; t.h[1] = h1[j]; t.h[2] = h2[j]; t.h[3] = h3[j];
        *(uint2*)(&Bs[nb + j][kb]) = t.u;
      }
    }
    ASYNC_WAIT();
    __syncthreads();

    // ---- compute: 16 WMMAs per wave per stage ----
#pragma unroll
    for (int ks = 0; ks < 2; ks++) {
      v16h a0 = frag_a_lds(&As[0][0], wm, 72, lane, ks * 32);
      v16h a1 = frag_a_lds(&As[0][0], wm + 16, 72, lane, ks * 32);
#pragma unroll
      for (int ni = 0; ni < 4; ni++) {
        v16h b = frag_b_lds(&Bs[0][0], wn + ni * 16, 72, lane, ks * 32);
        acc[0][ni] = wmma16x16x32(a0, b, acc[0][ni]);
        acc[1][ni] = wmma16x16x32(a1, b, acc[1][ni]);
      }
    }
  }

  const int hl = lane >> 4, ln = lane & 15;
#pragma unroll
  for (int mi = 0; mi < 2; mi++) {
#pragma unroll
    for (int ni = 0; ni < 4; ni++) {
      int coln = bn + wn + ni * 16 + ln;
      float bv = bias[coln];
#pragma unroll
      for (int r = 0; r < 8; r++) {
        int rowm = bm + wm + mi * 16 + r + 8 * hl;  // C layout: M = r + 8*hl
        float val = acc[mi][ni][r] + bv;
        if (mode == 0) {
          outF[(size_t)rowm * Ndim + coln] = val;
        } else if (mode == 1) {
          outH[(size_t)rowm * Ndim + coln] = (half_t)gelu_f(val);
        } else {  // QKV scatter into [B,NH,S,HD]; q scaled by 1/sqrt(HD)
          int part = coln >> 11;
          int head = (coln >> 6) & 31;
          int dd = coln & 63;
          int bb = rowm >> 10;
          int ss = rowm & 1023;
          size_t di = (((size_t)bb * NHn + head) * Sn + ss) * HDn + dd;
          half_t* dst = (part == 0) ? qo : ((part == 1) ? ko : vo);
          float scl = (part == 0) ? 0.125f : 1.0f;
          dst[di] = (half_t)(val * scl);
        }
      }
    }
  }
}

// ---------------------------------------------------------------------------
// Flash-style causal attention. grid=(S/64, B*NH), block=128 (4 waves x 16 q).
// q/k/v in [B,NH,S,HD] f16 (q pre-scaled). ctx out in [B,S,H] f16.
// ---------------------------------------------------------------------------
__global__ __launch_bounds__(128) void flash_attn_kernel(
    const half_t* __restrict__ Q, const half_t* __restrict__ K,
    const half_t* __restrict__ V, half_t* __restrict__ ctx) {
  __shared__ __align__(16) half_t Klds[32][72];  // [key][hd]
  __shared__ __align__(16) half_t Vt[64][40];    // [hd][key] transposed
  __shared__ float sc[4][16][33];                // wave-private scores
  __shared__ __align__(16) half_t pr[4][16][40]; // wave-private probs (A frag)
  __shared__ float smax[4][16], ssum[4][16], sfac[4][16];

  const int tid = threadIdx.x;
  const int lane = tid & 31;
  const int w = tid >> 5;
  const int qblk = blockIdx.x;            // 0..15
  const int bh = blockIdx.y;              // 0..127
  const int bb = bh >> 5, hh = bh & 31;
  const size_t base = (size_t)bh * Sn * HDn;
  const int qbase = qblk * 64 + w * 16;
  const int hl = lane >> 4, ln = lane & 15;

  // Q fragments for this wave's 16 rows (hd 0..31 and 32..63)
  v16h aq0, aq1;
  {
    const half_t* qrow = Q + base + (size_t)(qbase + ln) * HDn;
#pragma unroll
    for (int e = 0; e < 16; e++) {
      int k = (e & 7) + 8 * hl + 16 * (e >> 3);
      aq0[e] = qrow[k];
      aq1[e] = qrow[k + 32];
    }
  }

  v8f acc[4] = {};
  if (lane < 16) { smax[w][lane] = -3.0e38f; ssum[w][lane] = 0.f; }

  const int kend = qblk * 64 + 64;  // uniform across block -> barriers safe
  for (int j0 = 0; j0 < kend; j0 += 32) {
    __syncthreads();
    // ---- K chunk (32 keys x 64 hd), row-major ----
#if USE_ASYNC_LDS
#pragma unroll
    for (int i = 0; i < 2; i++) {
      int idx = i * 128 + tid;        // 256 x 16B chunks
      int kr = idx >> 3;              // key 0..31
      int c8 = (idx & 7) << 3;        // hd 0..56
      ASYNC_CP_B128(K + base + (size_t)(j0 + kr) * HDn + c8, &Klds[kr][c8]);
    }
#else
#pragma unroll
    for (int i = 0; i < 4; i++) {
      int idx = i * 128 + tid;        // 512 x uint2
      int kr = idx >> 4;
      int c4 = (idx & 15) << 2;
      uint2 dk = *(const uint2*)(K + base + (size_t)(j0 + kr) * HDn + c4);
      *(uint2*)(&Klds[kr][c4]) = dk;
    }
#endif
    // ---- V chunk -> Vt[hd][key] via 4x4 register transpose (1 sub-block/thr) --
    {
      int kb = (tid >> 4) << 2;       // key 0..28
      int nb = (tid & 15) << 2;       // hd 0..60
      uint2 d0 = *(const uint2*)(V + base + (size_t)(j0 + kb + 0) * HDn + nb);
      uint2 d1 = *(const uint2*)(V + base + (size_t)(j0 + kb + 1) * HDn + nb);
      uint2 d2 = *(const uint2*)(V + base + (size_t)(j0 + kb + 2) * HDn + nb);
      uint2 d3 = *(const uint2*)(V + base + (size_t)(j0 + kb + 3) * HDn + nb);
      const half_t* h0 = (const half_t*)&d0;
      const half_t* h1 = (const half_t*)&d1;
      const half_t* h2 = (const half_t*)&d2;
      const half_t* h3 = (const half_t*)&d3;
#pragma unroll
      for (int j = 0; j < 4; j++) {
        U64h t;
        t.h[0] = h0[j]; t.h[1] = h1[j]; t.h[2] = h2[j]; t.h[3] = h3[j];
        *(uint2*)(&Vt[nb + j][kb]) = t.u;
      }
    }
    ASYNC_WAIT();
    __syncthreads();

    // ---- scores (2 x 16-key tiles), causal mask, into wave-private LDS ----
#pragma unroll
    for (int nt = 0; nt < 2; nt++) {
      v8f c = {};
      v16h b0 = frag_b_lds(&Klds[0][0], nt * 16, 72, lane, 0);
      v16h b1 = frag_b_lds(&Klds[0][0], nt * 16, 72, lane, 32);
      c = wmma16x16x32(aq0, b0, c);
      c = wmma16x16x32(aq1, b1, c);
      int key = j0 + nt * 16 + ln;
#pragma unroll
      for (int r = 0; r < 8; r++) {
        int qm = r + 8 * hl;
        sc[w][qm][nt * 16 + ln] = (key <= qbase + qm) ? c[r] : -10000.0f;
      }
    }

    // ---- online softmax (2 lanes per row; same-wave LDS is in-order) ----
    {
      int row = ln, half = hl;
      float mx = -3.0e38f;
#pragma unroll
      for (int i = 0; i < 16; i++) mx = fmaxf(mx, sc[w][row][half * 16 + i]);
      float cmax = fmaxf(mx, __shfl_xor(mx, 16));
      float mold = smax[w][row];
      float mnew = fmaxf(mold, cmax);
      float f = __expf(mold - mnew);
      float ls = 0.f;
#pragma unroll
      for (int i = 0; i < 16; i++) {
        float p = __expf(sc[w][row][half * 16 + i] - mnew);
        pr[w][row][half * 16 + i] = (half_t)p;
        ls += p;
      }
      float lsum = ls + __shfl_xor(ls, 16);
      smax[w][row] = mnew;                       // both halves write same value
      ssum[w][row] = ssum[w][row] * f + lsum;
      sfac[w][row] = f;
    }

    // ---- rescale accumulators and add probs @ V ----
    float fr[8];
#pragma unroll
    for (int r = 0; r < 8; r++) fr[r] = sfac[w][r + 8 * hl];
#pragma unroll
    for (int nt = 0; nt < 4; nt++)
#pragma unroll
      for (int r = 0; r < 8; r++) acc[nt][r] *= fr[r];

    v16h ap = frag_a_lds(&pr[w][0][0], 0, 40, lane, 0);
#pragma unroll
    for (int nt = 0; nt < 4; nt++) {
      v16h bv = frag_b_lds(&Vt[0][0], nt * 16, 40, lane, 0);
      acc[nt] = wmma16x16x32(ap, bv, acc[nt]);
    }
  }

  // ---- normalize and write ctx[B,S,H] ----
  float inv[8];
#pragma unroll
  for (int r = 0; r < 8; r++) inv[r] = 1.0f / ssum[w][r + 8 * hl];
#pragma unroll
  for (int nt = 0; nt < 4; nt++)
#pragma unroll
    for (int r = 0; r < 8; r++) {
      int qabs = qbase + r + 8 * hl;
      size_t di = ((size_t)bb * Sn + qabs) * Hn + hh * HDn + nt * 16 + ln;
      ctx[di] = (half_t)(acc[nt][r] * inv[r]);
    }
}

// ---------------------------------------------------------------------------
// Launch
// ---------------------------------------------------------------------------
extern "C" void kernel_launch(void* const* d_in, const int* in_sizes, int n_in,
                              void* d_out, int out_size, void* d_ws, size_t ws_size,
                              hipStream_t stream) {
  const float* hidden  = (const float*)d_in[0];
  // d_in[1] = mask (causal tril) -- implemented analytically
  const float* ln1_w   = (const float*)d_in[2];
  const float* ln1_b   = (const float*)d_in[3];
  const float* w_qkv   = (const float*)d_in[4];
  const float* b_qkv   = (const float*)d_in[5];
  const float* w_dense = (const float*)d_in[6];
  const float* b_dense = (const float*)d_in[7];
  const float* ln3_w   = (const float*)d_in[8];
  const float* ln3_b   = (const float*)d_in[9];
  const float* ln2_w   = (const float*)d_in[10];
  const float* ln2_b   = (const float*)d_in[11];
  const float* w_h4h   = (const float*)d_in[12];
  const float* b_h4h   = (const float*)d_in[13];
  const float* w_4hh   = (const float*)d_in[14];
  const float* b_4hh   = (const float*)d_in[15];
  const float* ln4_w   = (const float*)d_in[16];
  const float* ln4_b   = (const float*)d_in[17];
  (void)in_sizes; (void)n_in; (void)out_size; (void)ws_size;

  char* ws = (char*)d_ws;
  size_t o = 0;
  half_t* wqkv_h   = (half_t*)(ws + o); o += (size_t)Hn * H3n * 2;
  half_t* wdense_h = (half_t*)(ws + o); o += (size_t)Hn * Hn * 2;
  half_t* wh4h_h   = (half_t*)(ws + o); o += (size_t)Hn * H4n * 2;
  half_t* w4hh_h   = (half_t*)(ws + o); o += (size_t)H4n * Hn * 2;
  half_t* x1_h     = (half_t*)(ws + o); o += (size_t)Mn * Hn * 2;  // reused as x2
  half_t* q_h      = (half_t*)(ws + o); o += (size_t)Mn * Hn * 2;
  half_t* k_h      = (half_t*)(ws + o); o += (size_t)Mn * Hn * 2;
  half_t* v_h      = (half_t*)(ws + o); o += (size_t)Mn * Hn * 2;
  half_t* ctx_h    = (half_t*)(ws + o); o += (size_t)Mn * Hn * 2;
  float*  attn_f   = (float*)(ws + o);  o += (size_t)Mn * Hn * 4;  // reused as mlp
  float*  lnin_f   = (float*)(ws + o);  o += (size_t)Mn * Hn * 4;
  half_t* inter_h  = (half_t*)(ws + o); o += (size_t)Mn * H4n * 2;
  half_t* x2_h = x1_h;
  float*  mlp_f = attn_f;

  // weights -> f16
  cvt_f32_f16<<<2048, 256, 0, stream>>>(w_qkv,   wqkv_h,   Hn * H3n);
  cvt_f32_f16<<<2048, 256, 0, stream>>>(w_dense, wdense_h, Hn * Hn);
  cvt_f32_f16<<<2048, 256, 0, stream>>>(w_h4h,   wh4h_h,   Hn * H4n);
  cvt_f32_f16<<<2048, 256, 0, stream>>>(w_4hh,   w4hh_h,   H4n * Hn);

  // LN1
  ln_to_f16<<<Mn, 256, 0, stream>>>(hidden, ln1_w, ln1_b, x1_h);

  // QKV GEMM (scatter epilogue, q pre-scaled)
  gemm_f16_kernel<<<dim3(H3n / 128, Mn / 128), 256, 0, stream>>>(
      x1_h, wqkv_h, b_qkv, Hn, H3n, 2, nullptr, nullptr, q_h, k_h, v_h);

  // causal flash attention
  flash_attn_kernel<<<dim3(Sn / 64, Bn * NHn), 128, 0, stream>>>(q_h, k_h, v_h, ctx_h);

  // attention dense projection
  gemm_f16_kernel<<<dim3(Hn / 128, Mn / 128), 256, 0, stream>>>(
      ctx_h, wdense_h, b_dense, Hn, Hn, 0, attn_f, nullptr, nullptr, nullptr, nullptr);

  // LN3 + residual + LN2
  fuse_ln3_res_ln2<<<Mn, 256, 0, stream>>>(attn_f, hidden, ln3_w, ln3_b, ln2_w, ln2_b,
                                           lnin_f, x2_h);

  // MLP h -> 4h with fused GELU
  gemm_f16_kernel<<<dim3(H4n / 128, Mn / 128), 256, 0, stream>>>(
      x2_h, wh4h_h, b_h4h, Hn, H4n, 1, nullptr, inter_h, nullptr, nullptr, nullptr);

  // MLP 4h -> h
  gemm_f16_kernel<<<dim3(Hn / 128, Mn / 128), 256, 0, stream>>>(
      inter_h, w4hh_h, b_4hh, H4n, Hn, 0, mlp_f, nullptr, nullptr, nullptr, nullptr);

  // LN4 + residual -> output
  final_ln4_res<<<Mn, 256, 0, stream>>>(mlp_f, lnin_f, ln4_w, ln4_b, (float*)d_out);
}
